// RelativeAttention_20624432955785
// MI455X (gfx1250) — compile-verified, matrix operand order checked by reading, not run on hardware
//
#include <hip/hip_runtime.h>
#include <hip/hip_bf16.h>

typedef _Float16 f16;
typedef _Float16 v16h __attribute__((ext_vector_type(16)));
typedef _Float16 v8h  __attribute__((ext_vector_type(8)));
typedef float    v8f  __attribute__((ext_vector_type(8)));

#define QLEN   512
#define MLEN   512
#define BATCH  4
#define DMODEL 1024
#define NHEAD  16
#define DHEAD  64
#define KLEN   1024
#define RLEN   1536
#define NPAIR  64            /* BATCH * NHEAD */
#define SCALE_F 0.125f       /* 1/sqrt(64) */
#define LN_EPS  1e-3f

union AF { v16h v; v8h h2[2]; f16 e[16]; };
union CF { v8f  v; float e[8]; };

#define WMMA_F16(a, bfr, c) __builtin_amdgcn_wmma_f32_16x16x32_f16( \
    false, (a).v, false, (bfr).v, (short)0, (c).v, false, false)

// ---------------------------------------------------------------------------
// Tiled WMMA GEMM:  C[M,N](f32) = A[M,K](f16) * B^T, with B stored [N][K] f16.
// Double-buffered LDS; B staged fragment-major (b128-only LDS traffic).
// ---------------------------------------------------------------------------
template<int BM, int BN, int WM, int WN>
__global__ __launch_bounds__(256)
void wmma_gemm(const f16* __restrict__ A, const f16* __restrict__ B,
               float* __restrict__ C, int M, int N, int K,
               long long sA, long long sB, long long sC)
{
    constexpr int BK = 32;
    constexpr int WAVES_N = BN / WN;
    constexpr int NT = BN / 16;
    constexpr int ALD = BK + 8;
    constexpr int BSLOT = 24;
    constexpr int A_VECS = (BM * BK) / (256 * 8);
    constexpr int B_VECS = (BN * BK) / (256 * 8);
    static_assert((BM / WM) * WAVES_N == 8, "8 waves per block");

    __shared__ __align__(16) f16 ldsA[2][BM * ALD];
    __shared__ __align__(16) f16 ldsB[2][NT * 32 * BSLOT];

    const int bm0 = blockIdx.y * BM;
    const int bn0 = blockIdx.x * BN;
    A += (long long)blockIdx.z * sA + (long long)bm0 * K;
    B += (long long)blockIdx.z * sB + (long long)bn0 * K;
    C += (long long)blockIdx.z * sC;

    const int tid  = threadIdx.x;
    const int wave = tid >> 5;
    const int lane = tid & 31;
    const int half = lane >> 4;
    const int l16  = lane & 15;
    const int wm   = wave / WAVES_N;
    const int wn   = wave % WAVES_N;

    int aRow[A_VECS], aCol[A_VECS];
    #pragma unroll
    for (int c = 0; c < A_VECS; ++c) {
        int idx = (c * 256 + tid) * 8;
        aRow[c] = idx / BK;
        aCol[c] = idx % BK;
    }
    int bRow[B_VECS], bCol[B_VECS], bDst[B_VECS];
    #pragma unroll
    for (int c = 0; c < B_VECS; ++c) {
        int idx = (c * 256 + tid) * 8;
        int nn = idx / BK;
        int cc = idx % BK;
        bRow[c] = nn;
        bCol[c] = cc;
        int t  = nn >> 4;
        int bl = (nn & 15) + ((cc >> 3) & 1) * 16;
        int be = (cc & 16) >> 1;
        bDst[c] = (t * 32 + bl) * BSLOT + be;
    }

    const int kTiles = K / BK;
    v8h regA[A_VECS], regB[B_VECS];

    auto stage = [&](int kt) {
        const int k0 = kt * BK;
        #pragma unroll
        for (int c = 0; c < A_VECS; ++c)
            regA[c] = *(const v8h*)&A[(long long)aRow[c] * K + k0 + aCol[c]];
        #pragma unroll
        for (int c = 0; c < B_VECS; ++c)
            regB[c] = *(const v8h*)&B[(long long)bRow[c] * K + k0 + bCol[c]];
        if (kt + 1 < kTiles) {
            __builtin_prefetch(&A[(long long)aRow[0] * K + (kt + 1) * BK + aCol[0]], 0, 1);
            __builtin_prefetch(&B[(long long)bRow[0] * K + (kt + 1) * BK + bCol[0]], 0, 1);
        }
    };
    auto commit = [&](int buf) {
        #pragma unroll
        for (int c = 0; c < A_VECS; ++c)
            *(v8h*)&ldsA[buf][aRow[c] * ALD + aCol[c]] = regA[c];
        #pragma unroll
        for (int c = 0; c < B_VECS; ++c)
            *(v8h*)&ldsB[buf][bDst[c]] = regB[c];
    };

    CF acc[WM / 16][WN / 16];
    #pragma unroll
    for (int mi = 0; mi < WM / 16; ++mi)
        #pragma unroll
        for (int ni = 0; ni < WN / 16; ++ni)
            #pragma unroll
            for (int e = 0; e < 8; ++e) acc[mi][ni].e[e] = 0.0f;

    stage(0);
    commit(0);
    __syncthreads();

    for (int kt = 0; kt < kTiles; ++kt) {
        if (kt + 1 < kTiles) stage(kt + 1);

        AF af[WM / 16], bf[WN / 16];
        #pragma unroll
        for (int mi = 0; mi < WM / 16; ++mi) {
            const f16* base = &ldsA[kt & 1][(wm * WM + mi * 16 + l16) * ALD];
            af[mi].h2[0] = *(const v8h*)(base + half * 8);
            af[mi].h2[1] = *(const v8h*)(base + 16 + half * 8);
        }
        #pragma unroll
        for (int ni = 0; ni < WN / 16; ++ni) {
            const f16* bp = &ldsB[kt & 1][((wn * (WN / 16) + ni) * 32 + lane) * BSLOT];
            bf[ni].h2[0] = *(const v8h*)bp;
            bf[ni].h2[1] = *(const v8h*)(bp + 8);
        }
        #pragma unroll
        for (int mi = 0; mi < WM / 16; ++mi)
            #pragma unroll
            for (int ni = 0; ni < WN / 16; ++ni)
                acc[mi][ni].v = WMMA_F16(af[mi], bf[ni], acc[mi][ni]);
        __syncthreads();
        if (kt + 1 < kTiles) {
            commit((kt + 1) & 1);
            __syncthreads();
        }
    }

    #pragma unroll
    for (int mi = 0; mi < WM / 16; ++mi)
        #pragma unroll
        for (int ni = 0; ni < WN / 16; ++ni) {
            int row = bm0 + wm * WM + mi * 16 + half * 8;
            int col = bn0 + wn * WN + ni * 16 + l16;
            #pragma unroll
            for (int v = 0; v < 8; ++v)
                C[(long long)(row + v) * N + col] = acc[mi][ni].e[v];
        }
}

// ---------------------------------------------------------------------------
// Fused flash-style relative attention (BI=32 q-rows, BJ=64 j-tiles).
// Register-pipelined staging: next tile's global loads issue before this
// tile's WMMA phases; LDS commit happens after the end-of-tile barrier.
// ---------------------------------------------------------------------------
#define BI 32
#define BJ 64
#define NJT (KLEN / BJ)

__global__ __launch_bounds__(256)
void flash_rel_attn_kernel(const f16* __restrict__ qw16, const f16* __restrict__ qr16,
                           const f16* __restrict__ k16p, const f16* __restrict__ kr16p,
                           const f16* __restrict__ vT16p, const float* __restrict__ ef1,
                           const unsigned char* __restrict__ selb,
                           f16* __restrict__ av16)
{
    const int i0   = blockIdx.x * BI;
    const int pair = blockIdx.y;
    const int b = pair >> 4;
    const int n = pair & 15;

    __shared__ __align__(16) f16 sQW[BI * 72];
    __shared__ __align__(16) f16 sQR[BI * 72];
    __shared__ __align__(16) f16 sK [2 * 4 * 32 * 16];   // [kstep][nt][lane][16]
    __shared__ __align__(16) f16 sKR[2 * 8 * 32 * 16];   // 128-wide bd band
    __shared__ __align__(16) f16 sV [2 * 4 * 32 * 16];
    __shared__ __align__(16) f16 sW [BI * 136];          // raw bd band (f16)
    __shared__ __align__(16) f16 sP [BI * 72];           // probabilities tile
    __shared__ __align__(16) unsigned char sSel[BI * 80];
    __shared__ float sM[BI];
    __shared__ float sL[BI];
    __shared__ float sTmax[2][16][4];
    __shared__ float sTsum[2][16][4];

    const int tid  = threadIdx.x;
    const int wave = tid >> 5;
    const int lane = tid & 31;
    const int half = lane >> 4;
    const int l16  = lane & 15;
    const int mi   = wave >> 2;     // 0..1 : 16-row tile
    const int nt   = wave & 3;      // 0..3 : 16-col tile

    const f16* Qw  = qw16 + ((long long)pair * QLEN + i0) * DHEAD;
    const f16* Qr  = qr16 + ((long long)pair * QLEN + i0) * DHEAD;
    const f16* Kp  = k16p + (long long)pair * KLEN * DHEAD;
    const f16* Krp = kr16p + (long long)pair * RLEN * DHEAD;
    const f16* Vp  = vT16p + (long long)pair * DHEAD * KLEN;
    const unsigned char* Sb = selb + (long long)b * QLEN * KLEN;

    // ---- prologue: stage Qw/Qr rows, init m/l, hoist ef1 ----
    {
        int idx = tid * 8;                 // 32*64 halves -> 1 v8h per thread
        int rr = idx >> 6, cc = idx & 63;
        *(v8h*)&sQW[rr * 72 + cc] = *(const v8h*)&Qw[(long long)rr * DHEAD + cc];
        *(v8h*)&sQR[rr * 72 + cc] = *(const v8h*)&Qr[(long long)rr * DHEAD + cc];
    }
    if (tid < BI) { sM[tid] = -3.0e38f; sL[tid] = 0.0f; }

    float e0v[8], e1v[8];
    #pragma unroll
    for (int v = 0; v < 8; ++v) {
        int gi = i0 + mi * 16 + half * 8 + v;
        long long eb = ((long long)(gi * BATCH + b) * NHEAD + n) * 2;
        e0v[v] = ef1[eb];
        e1v[v] = ef1[eb + 1];
    }
    CF O;
    #pragma unroll
    for (int e = 0; e < 8; ++e) O.e[e] = 0.0f;

    // staging coordinates (j-tile invariant)
    int kNN[2], kCC[2], kDst[2];
    #pragma unroll
    for (int c = 0; c < 2; ++c) {
        int idx = (c * 256 + tid) * 8;
        int nn = idx >> 6, cc = idx & 63;
        int c2 = cc & 31;
        kNN[c] = nn; kCC[c] = cc;
        kDst[c] = ((((cc >> 5) * 4 + (nn >> 4)) * 32 + (nn & 15) + ((c2 >> 3) & 1) * 16) << 4)
                  + ((c2 & 16) >> 1);
    }
    int rNN[4], rCC[4], rDst[4];
    #pragma unroll
    for (int c = 0; c < 4; ++c) {
        int idx = (c * 256 + tid) * 8;
        int nn = idx >> 6, cc = idx & 63;
        int c2 = cc & 31;
        rNN[c] = nn; rCC[c] = cc;
        rDst[c] = ((((cc >> 5) * 8 + (nn >> 4)) * 32 + (nn & 15) + ((c2 >> 3) & 1) * 16) << 4)
                  + ((c2 & 16) >> 1);
    }
    const bool selAct = tid < 128;
    const int selRR = tid >> 2, selCG = (tid & 3) * 16;

    v8h rgK[2], rgV[2], rgKR[4];
    uint4 rgSel;

    auto stageRegs = [&](int jt2) {
        const int j0n = jt2 * BJ;
        const int jb2 = j0n + QLEN - i0 - (BI - 1);
        #pragma unroll
        for (int c = 0; c < 2; ++c) {
            rgK[c] = *(const v8h*)&Kp[(long long)(j0n + kNN[c]) * DHEAD + kCC[c]];
            rgV[c] = *(const v8h*)&Vp[(long long)kNN[c] * KLEN + j0n + kCC[c]];
        }
        #pragma unroll
        for (int c = 0; c < 4; ++c) {
            int row = jb2 + rNN[c];
            if (row > RLEN - 1) row = RLEN - 1;
            rgKR[c] = *(const v8h*)&Krp[(long long)row * DHEAD + rCC[c]];
        }
        if (selAct)
            rgSel = *(const uint4*)&Sb[(long long)(i0 + selRR) * KLEN + j0n + selCG];
        if (jt2 + 1 < NJT) {   // prefetch next tile (global_prefetch_b8)
            __builtin_prefetch(&Kp[(long long)(j0n + BJ + kNN[0]) * DHEAD + kCC[0]], 0, 1);
            __builtin_prefetch(&Vp[(long long)kNN[0] * KLEN + j0n + BJ + kCC[0]], 0, 1);
        }
    };
    auto commitRegs = [&]() {
        #pragma unroll
        for (int c = 0; c < 2; ++c) {
            *(v8h*)&sK[kDst[c]] = rgK[c];
            *(v8h*)&sV[kDst[c]] = rgV[c];
        }
        #pragma unroll
        for (int c = 0; c < 4; ++c)
            *(v8h*)&sKR[rDst[c]] = rgKR[c];
        if (selAct)
            *(uint4*)&sSel[selRR * 80 + selCG] = rgSel;
    };

    auto loadA = [&](const f16* rowbase, int kst, AF& f) {
        f.h2[0] = *(const v8h*)(rowbase + kst * 32 + half * 8);
        f.h2[1] = *(const v8h*)(rowbase + kst * 32 + 16 + half * 8);
    };
    auto loadB = [&](const f16* arr, int NT_, int kst, int t, AF& f) {
        const f16* p = arr + (((kst * NT_ + t) * 32 + lane) << 4);
        f.h2[0] = *(const v8h*)p;
        f.h2[1] = *(const v8h*)(p + 8);
    };

    stageRegs(0);
    __syncthreads();   // Qw/Qr + m/l init visible

    for (int jt = 0; jt < NJT; ++jt) {
        const int jj = nt * 16 + l16;

        commitRegs();
        __syncthreads();   // S1: tile visible in LDS
        if (jt + 1 < NJT) stageRegs(jt + 1);   // overlap with WMMA phases

        // ---- bd band: W = Qr x Kr^T  (2x8 tiles; wave -> mi row, 2 col tiles) ----
        {
            CF wacc[2];
            #pragma unroll
            for (int wi = 0; wi < 2; ++wi)
                #pragma unroll
                for (int e = 0; e < 8; ++e) wacc[wi].e[e] = 0.0f;
            AF a0, a1, bfr;
            const f16* arow = &sQR[(mi * 16 + l16) * 72];
            loadA(arow, 0, a0);
            loadA(arow, 1, a1);
            #pragma unroll
            for (int wi = 0; wi < 2; ++wi) {
                int t = nt * 2 + wi;
                loadB(sKR, 8, 0, t, bfr);
                wacc[wi].v = WMMA_F16(a0, bfr, wacc[wi]);
                loadB(sKR, 8, 1, t, bfr);
                wacc[wi].v = WMMA_F16(a1, bfr, wacc[wi]);
            }
            #pragma unroll
            for (int wi = 0; wi < 2; ++wi) {
                int colb = (nt * 2 + wi) * 16 + l16;
                #pragma unroll
                for (int v = 0; v < 8; ++v) {
                    int ii = mi * 16 + half * 8 + v;
                    sW[ii * 136 + colb] = (f16)wacc[wi].e[v];
                }
            }
        }
        __syncthreads();   // S2

        // ---- scores: S = Qw x K^T (one 16x16 tile per wave) + epilogue ----
        CF sacc;
        #pragma unroll
        for (int e = 0; e < 8; ++e) sacc.e[e] = 0.0f;
        {
            AF a0, bfr;
            const f16* arow = &sQW[(mi * 16 + l16) * 72];
            loadA(arow, 0, a0);
            loadB(sK, 4, 0, nt, bfr);
            sacc.v = WMMA_F16(a0, bfr, sacc);
            loadA(arow, 1, a0);
            loadB(sK, 4, 1, nt, bfr);
            sacc.v = WMMA_F16(a0, bfr, sacc);
        }
        float sv[8];
        #pragma unroll
        for (int v = 0; v < 8; ++v) {
            int ii = mi * 16 + half * 8 + v;
            float bdv = (float)sW[ii * 136 + (jj + (BI - 1) - ii)];
            unsigned char sm = sSel[ii * 80 + jj];
            float ef = (sm & 1) ? e1v[v] : e0v[v];
            float s = (sacc.e[v] + bdv + ef) * SCALE_F;
            if (sm & 2) s -= 1.0e30f;
            sv[v] = s;
        }
        float rmax[8];
        #pragma unroll
        for (int v = 0; v < 8; ++v) {
            float m = sv[v];
            m = fmaxf(m, __shfl_xor(m, 1));
            m = fmaxf(m, __shfl_xor(m, 2));
            m = fmaxf(m, __shfl_xor(m, 4));
            m = fmaxf(m, __shfl_xor(m, 8));
            rmax[v] = m;
        }
        if (l16 == 0) {
            #pragma unroll
            for (int v = 0; v < 8; ++v) sTmax[mi][half * 8 + v][nt] = rmax[v];
        }
        __syncthreads();   // S3

        float mnew[8], alpha[8], rsum[8];
        #pragma unroll
        for (int v = 0; v < 8; ++v) {
            int rr = half * 8 + v;
            float tm = fmaxf(fmaxf(sTmax[mi][rr][0], sTmax[mi][rr][1]),
                             fmaxf(sTmax[mi][rr][2], sTmax[mi][rr][3]));
            int ii = mi * 16 + rr;
            float mo = sM[ii];
            float mn = fmaxf(mo, tm);
            mnew[v]  = mn;
            alpha[v] = __expf(mo - mn);
            float p  = __expf(sv[v] - mn);
            sv[v] = p;
            float su = p;
            su += __shfl_xor(su, 1);
            su += __shfl_xor(su, 2);
            su += __shfl_xor(su, 4);
            su += __shfl_xor(su, 8);
            rsum[v] = su;
        }
        if (l16 == 0) {
            #pragma unroll
            for (int v = 0; v < 8; ++v) sTsum[mi][half * 8 + v][nt] = rsum[v];
        }
        #pragma unroll
        for (int v = 0; v < 8; ++v) {
            int ii = mi * 16 + half * 8 + v;
            sP[ii * 72 + jj] = (f16)sv[v];
        }
        #pragma unroll
        for (int v = 0; v < 8; ++v) O.e[v] *= alpha[v];
        __syncthreads();   // S4

        if (nt == 0 && l16 == 0) {
            #pragma unroll
            for (int v = 0; v < 8; ++v) {
                int rr = half * 8 + v;
                int ii = mi * 16 + rr;
                float ts = sTsum[mi][rr][0] + sTsum[mi][rr][1] +
                           sTsum[mi][rr][2] + sTsum[mi][rr][3];
                sL[ii] = sL[ii] * alpha[v] + ts;
                sM[ii] = mnew[v];
            }
        }

        // ---- PV: O += P x V  (one 16x16 out tile per wave) ----
        {
            AF a0, bfr;
            const f16* prow = &sP[(mi * 16 + l16) * 72];
            loadA(prow, 0, a0);
            loadB(sV, 4, 0, nt, bfr);
            O.v = WMMA_F16(a0, bfr, O);
            loadA(prow, 1, a0);
            loadB(sV, 4, 1, nt, bfr);
            O.v = WMMA_F16(a0, bfr, O);
        }
        __syncthreads();   // S5: LDS free for next commit
    }

    float lv[8];
    #pragma unroll
    for (int v = 0; v < 8; ++v) lv[v] = sL[mi * 16 + half * 8 + v];
    #pragma unroll
    for (int v = 0; v < 8; ++v) {
        int gi = i0 + mi * 16 + half * 8 + v;
        float o = O.e[v] / lv[v];
        av16[(long long)(gi * BATCH + b) * DMODEL + n * DHEAD + nt * 16 + l16] = (f16)o;
    }
}

// ---------------------------------------------------------------------------
// Elementwise / pack kernels
// ---------------------------------------------------------------------------
__global__ void cvt_f16_kernel(const float* __restrict__ src,
                               f16* __restrict__ dst, long long n)
{
    long long t = (long long)blockIdx.x * blockDim.x + threadIdx.x;
    if (t < n) dst[t] = (f16)src[t];
}

__global__ void cvt_transpose_kernel(const float* __restrict__ src, f16* __restrict__ dst)
{
    long long t = (long long)blockIdx.x * blockDim.x + threadIdx.x;
    if (t >= (long long)DMODEL * DMODEL) return;
    int cc = (int)(t % DMODEL);
    int rr = (int)(t / DMODEL);
    dst[(long long)cc * DMODEL + rr] = (f16)src[t];
}

__global__ void pack_q_kernel(const float* __restrict__ qh,
                              const float* __restrict__ rwb,
                              const float* __restrict__ rrb,
                              f16* __restrict__ qw, f16* __restrict__ qr)
{
    long long t = (long long)blockIdx.x * blockDim.x + threadIdx.x;
    if (t >= (long long)QLEN * BATCH * DMODEL) return;
    int col = (int)(t % DMODEL);
    int row = (int)(t / DMODEL);
    int i = row >> 2, b = row & 3;
    int n = col >> 6, d = col & 63;
    int pair = b * NHEAD + n;
    float q = qh[t];
    long long o = ((long long)pair * QLEN + i) * DHEAD + d;
    qw[o] = (f16)(q + rwb[col]);
    qr[o] = (f16)(q + rrb[col]);
}

__global__ void pack_kv_kernel(const float* __restrict__ src,
                               f16* __restrict__ dst, int jlen)
{
    long long t = (long long)blockIdx.x * blockDim.x + threadIdx.x;
    if (t >= (long long)jlen * BATCH * DMODEL) return;
    int col = (int)(t % DMODEL);
    long long row = t / DMODEL;
    int j = (int)(row >> 2), b = (int)(row & 3);
    int n = col >> 6, d = col & 63;
    int pair = b * NHEAD + n;
    dst[((long long)pair * jlen + j) * DHEAD + d] = (f16)src[t];
}

__global__ void pack_vT_kernel(const float* __restrict__ src, f16* __restrict__ dst)
{
    long long t = (long long)blockIdx.x * blockDim.x + threadIdx.x;
    if (t >= (long long)KLEN * BATCH * DMODEL) return;
    int col = (int)(t % DMODEL);
    long long row = t / DMODEL;
    int j = (int)(row >> 2), b = (int)(row & 3);
    int n = col >> 6, d = col & 63;
    int pair = b * NHEAD + n;
    dst[((long long)pair * DHEAD + d) * KLEN + j] = (f16)src[t];
}

__global__ void ef1_kernel(const float* __restrict__ qh,
                           const float* __restrict__ rsb,
                           const float* __restrict__ seg_embed,
                           float* __restrict__ ef1)
{
    int t = blockIdx.x * blockDim.x + threadIdx.x;
    if (t >= QLEN * BATCH * NHEAD * 2) return;
    int s = t & 1;
    int n = (t >> 1) & 15;
    int b = (t >> 5) & 3;
    int i = t >> 7;
    const float* q  = qh + (long long)(i * BATCH + b) * DMODEL + n * DHEAD;
    const float* bi = rsb + n * DHEAD;
    const float* se = seg_embed + (s * NHEAD + n) * DHEAD;
    float acc = 0.0f;
    for (int d = 0; d < DHEAD; ++d) acc += (q[d] + bi[d]) * se[d];
    ef1[t] = acc;
}

__global__ void selmask_kernel(const float* __restrict__ seg_mat,
                               const float* __restrict__ attn_mask,
                               unsigned char* __restrict__ selb)
{
    long long t = (long long)blockIdx.x * blockDim.x + threadIdx.x;
    if (t >= (long long)BATCH * QLEN * KLEN) return;
    int j = (int)(t % KLEN);
    long long r = t / KLEN;
    int i = (int)(r % QLEN);
    int b = (int)(r / QLEN);
    long long smi = (long long)(i * KLEN + j) * BATCH + b;
    unsigned char s = (unsigned char)((seg_mat[smi * 2 + 1] > 0.5f ? 1 : 0) |
                                      (attn_mask[smi] > 0.5f ? 2 : 0));
    selb[t] = s;
}

__global__ __launch_bounds__(256)
void ln_kernel(const float* __restrict__ attn_out, const float* __restrict__ h,
               const float* __restrict__ gamma, const float* __restrict__ beta,
               float* __restrict__ out)
{
    const int row = blockIdx.x;
    const int tid = threadIdx.x;
    __shared__ float red[256];
    const float* a  = attn_out + (long long)row * DMODEL;
    const float* hh = h + (long long)row * DMODEL;

    float x[4];
    float sum = 0.0f;
    #pragma unroll
    for (int c = 0; c < 4; ++c) {
        int j = tid + c * 256;
        x[c] = a[j] + hh[j];
        sum += x[c];
    }
    red[tid] = sum; __syncthreads();
    for (int st = 128; st > 0; st >>= 1) {
        if (tid < st) red[tid] += red[tid + st];
        __syncthreads();
    }
    float mu = red[0] * (1.0f / DMODEL); __syncthreads();

    float vs = 0.0f;
    #pragma unroll
    for (int c = 0; c < 4; ++c) { float d = x[c] - mu; vs += d * d; }
    red[tid] = vs; __syncthreads();
    for (int st = 128; st > 0; st >>= 1) {
        if (tid < st) red[tid] += red[tid + st];
        __syncthreads();
    }
    float rinv = rsqrtf(red[0] * (1.0f / DMODEL) + LN_EPS);

    #pragma unroll
    for (int c = 0; c < 4; ++c) {
        int j = tid + c * 256;
        out[(long long)row * DMODEL + j] = (x[c] - mu) * rinv * gamma[j] + beta[j];
    }
}

// ---------------------------------------------------------------------------
// Host orchestration
// ---------------------------------------------------------------------------
static inline int ceil_div_ll(long long a, int b) { return (int)((a + b - 1) / b); }

extern "C" void kernel_launch(void* const* d_in, const int* in_sizes, int n_in,
                              void* d_out, int out_size, void* d_ws, size_t ws_size,
                              hipStream_t stream)
{
    (void)in_sizes; (void)n_in; (void)out_size; (void)ws_size;
    const float* r_w_bias = (const float*)d_in[0];
    const float* r_s_bias = (const float*)d_in[1];
    const float* r_r_bias = (const float*)d_in[2];
    const float* seg_embed = (const float*)d_in[3];
    const float* h    = (const float*)d_in[4];
    const float* r    = (const float*)d_in[5];
    const float* mems = (const float*)d_in[6];
    const float* seg_mat   = (const float*)d_in[7];
    const float* attn_mask = (const float*)d_in[8];
    const float* wq = (const float*)d_in[9];
    const float* wk = (const float*)d_in[10];
    const float* wv = (const float*)d_in[11];
    const float* wr = (const float*)d_in[12];
    const float* wo = (const float*)d_in[13];
    const float* ln_gamma = (const float*)d_in[14];
    const float* ln_beta  = (const float*)d_in[15];
    float* out = (float*)d_out;

    char* ws = (char*)d_ws;
    size_t off = 0;
    auto alloc = [&](size_t bytes) -> void* {
        off = (off + 255) & ~(size_t)255;
        void* p = ws + off;
        off += bytes;
        return p;
    };

    const long long nHB  = (long long)QLEN * BATCH * DMODEL;
    const long long nCAT = (long long)KLEN * BATCH * DMODEL;
    const long long nR   = (long long)RLEN * BATCH * DMODEL;
    const long long nW   = (long long)DMODEL * DMODEL;

    f16*   h16    = (f16*)alloc(nHB * 2);
    f16*   cat16  = (f16*)alloc(nCAT * 2);
    f16*   r16    = (f16*)alloc(nR * 2);
    f16*   wqT16  = (f16*)alloc(nW * 2);
    f16*   wkT16  = (f16*)alloc(nW * 2);
    f16*   wvT16  = (f16*)alloc(nW * 2);
    f16*   wrT16  = (f16*)alloc(nW * 2);
    f16*   wo16   = (f16*)alloc(nW * 2);
    float* qhead  = (float*)alloc(nHB * 4);
    float* khead  = (float*)alloc(nCAT * 4);
    float* vhead  = (float*)alloc(nCAT * 4);
    float* krh    = (float*)alloc(nR * 4);
    f16*   qw16   = (f16*)alloc((long long)NPAIR * QLEN * DHEAD * 2);
    f16*   qr16   = (f16*)alloc((long long)NPAIR * QLEN * DHEAD * 2);
    f16*   k16p   = (f16*)alloc((long long)NPAIR * KLEN * DHEAD * 2);
    f16*   vT16p  = (f16*)alloc((long long)NPAIR * DHEAD * KLEN * 2);
    f16*   kr16p  = (f16*)alloc((long long)NPAIR * RLEN * DHEAD * 2);
    float* ef1    = (float*)alloc((long long)QLEN * BATCH * NHEAD * 2 * 4);
    unsigned char* selb = (unsigned char*)alloc((long long)BATCH * QLEN * KLEN);
    f16*   av16   = (f16*)alloc(nHB * 2);
    float* aout   = (float*)alloc(nHB * 4);

    dim3 blk(256);

    cvt_f16_kernel<<<ceil_div_ll(nHB, 256), blk, 0, stream>>>(h, h16, nHB);
    cvt_f16_kernel<<<ceil_div_ll(nHB, 256), blk, 0, stream>>>(mems, cat16, nHB);
    cvt_f16_kernel<<<ceil_div_ll(nHB, 256), blk, 0, stream>>>(h, cat16 + nHB, nHB);
    cvt_f16_kernel<<<ceil_div_ll(nR, 256), blk, 0, stream>>>(r, r16, nR);
    cvt_transpose_kernel<<<ceil_div_ll(nW, 256), blk, 0, stream>>>(wq, wqT16);
    cvt_transpose_kernel<<<ceil_div_ll(nW, 256), blk, 0, stream>>>(wk, wkT16);
    cvt_transpose_kernel<<<ceil_div_ll(nW, 256), blk, 0, stream>>>(wv, wvT16);
    cvt_transpose_kernel<<<ceil_div_ll(nW, 256), blk, 0, stream>>>(wr, wrT16);
    cvt_f16_kernel<<<ceil_div_ll(nW, 256), blk, 0, stream>>>(wo, wo16, nW);
    selmask_kernel<<<ceil_div_ll((long long)BATCH * QLEN * KLEN, 256), blk, 0, stream>>>(
        seg_mat, attn_mask, selb);

    wmma_gemm<128, 128, 64, 32><<<dim3(DMODEL / 128, (QLEN * BATCH) / 128, 1), blk, 0, stream>>>(
        h16, wqT16, qhead, QLEN * BATCH, DMODEL, DMODEL, 0, 0, 0);
    wmma_gemm<128, 128, 64, 32><<<dim3(DMODEL / 128, (KLEN * BATCH) / 128, 1), blk, 0, stream>>>(
        cat16, wkT16, khead, KLEN * BATCH, DMODEL, DMODEL, 0, 0, 0);
    wmma_gemm<128, 128, 64, 32><<<dim3(DMODEL / 128, (KLEN * BATCH) / 128, 1), blk, 0, stream>>>(
        cat16, wvT16, vhead, KLEN * BATCH, DMODEL, DMODEL, 0, 0, 0);
    wmma_gemm<128, 128, 64, 32><<<dim3(DMODEL / 128, (RLEN * BATCH) / 128, 1), blk, 0, stream>>>(
        r16, wrT16, krh, RLEN * BATCH, DMODEL, DMODEL, 0, 0, 0);

    pack_q_kernel<<<ceil_div_ll(nHB, 256), blk, 0, stream>>>(
        qhead, r_w_bias, r_r_bias, qw16, qr16);
    pack_kv_kernel<<<ceil_div_ll(nCAT, 256), blk, 0, stream>>>(khead, k16p, KLEN);
    pack_vT_kernel<<<ceil_div_ll(nCAT, 256), blk, 0, stream>>>(vhead, vT16p);
    pack_kv_kernel<<<ceil_div_ll(nR, 256), blk, 0, stream>>>(krh, kr16p, RLEN);
    ef1_kernel<<<(QLEN * BATCH * NHEAD * 2) / 256, blk, 0, stream>>>(
        qhead, r_s_bias, seg_embed, ef1);

    flash_rel_attn_kernel<<<dim3(QLEN / BI, NPAIR), blk, 0, stream>>>(
        qw16, qr16, k16p, kr16p, vT16p, ef1, selb, av16);

    wmma_gemm<128, 128, 64, 32><<<dim3(DMODEL / 128, (QLEN * BATCH) / 128, 1), blk, 0, stream>>>(
        av16, wo16, aout, QLEN * BATCH, DMODEL, DMODEL, 0, 0, 0);
    ln_kernel<<<QLEN * BATCH, blk, 0, stream>>>(aout, h, ln_gamma, ln_beta, out);
}
